// GIN1Policy_84000970375206
// MI455X (gfx1250) — matrix operand order repaired and checked by reading, not compiled, not run on hardware
//
#include <hip/hip_runtime.h>
#include <math.h>

#define EMB 128
typedef __attribute__((ext_vector_type(16))) _Float16 v16h;
typedef __attribute__((ext_vector_type(8)))  _Float16 v8h;
typedef __attribute__((ext_vector_type(8)))  float    v8f;

#define ACT_NONE 0
#define ACT_RELU 1

// Pre-pack a weight matrix W[K,128] fp32 -> Wt[128,K] f16 (B-fragment layout:
// contiguous K per output column). Tiny one-shot kernel, L2-resident data.
__global__ __launch_bounds__(256) void pack_wt(
    const float* __restrict__ W, _Float16* __restrict__ Wt, int K)
{
  int t = blockIdx.x * 256 + threadIdx.x;
  if (t >= K * 128) return;
  int c = t / K;
  int k = t - c * K;
  Wt[(size_t)c * K + k] = (_Float16)W[(size_t)k * 128 + c];
}

// ---------------------------------------------------------------------------
// Tiled WMMA GEMM:  Y[row, colOff+col] = act( (alpha*X + X2)[row,:] @ W[K,128] + b[col] )
// X: [N,K] fp32 row-major; Wt: [128,K] f16 pre-transposed.
// Block = 64 rows x 128 cols, 256 threads (8 waves). Wave w owns col slab
// w*16..+15, iterates 4 row tiles -> 4 WMMAs per K-step on one B fragment.
// A staged once in LDS (stride K+24 halves: 16B-aligned rows, bank-spread);
// B fragments come straight from global (two 16B b128 loads, WGP$-resident).
// Single barrier per block, no staging inside the K loop.
// ---------------------------------------------------------------------------
__global__ __launch_bounds__(256) void gemm_wmma(
    const float* __restrict__ X, const float* __restrict__ X2,
    const float* __restrict__ epsPtr,
    const _Float16* __restrict__ Wt, const float* __restrict__ bias,
    float* __restrict__ Y, int N, int K, int ldY, int colOff, int act)
{
  __shared__ _Float16 lA[64 * 408];   // 64 rows x (K<=384)+24 pad

  const int tid  = threadIdx.x;
  const int lane = tid & 31;
  const int wave = tid >> 5;
  const int row0 = blockIdx.x * 64;
  const int Kpad = K + 24;

  float alpha = 1.0f;
  if (epsPtr) alpha = 1.0f + epsPtr[0];

  // Stage A tile (fp32 -> f16, fused (1+eps)*tgt + agg), float4-vectorized.
  const int Kq = K >> 2;              // K is a multiple of 4 (128 or 384)
  for (int i = tid; i < 64 * Kq; i += 256) {
    int r = i / Kq;
    int k = (i - r * Kq) << 2;
    int row = row0 + r;
    float4 v = make_float4(0.f, 0.f, 0.f, 0.f);
    if (row < N) {
      v = *(const float4*)(X + (size_t)row * K + k);
      v.x *= alpha; v.y *= alpha; v.z *= alpha; v.w *= alpha;
      if (X2) {
        const float4 u = *(const float4*)(X2 + (size_t)row * K + k);
        v.x += u.x; v.y += u.y; v.z += u.z; v.w += u.w;
      }
    }
    _Float16* dst = &lA[r * Kpad + k];
    dst[0] = (_Float16)v.x; dst[1] = (_Float16)v.y;
    dst[2] = (_Float16)v.z; dst[3] = (_Float16)v.w;
  }
  __syncthreads();

  v8f acc0 = {}, acc1 = {}, acc2 = {}, acc3 = {};
  const int col0 = wave * 16;
  const int colL = lane & 15;
  const int hiA  = (lane >> 4) * 8;    // A: lanes 16-31 hold K+8 / K+24 groups
  const int hiB  = (lane >> 4) * 8;    // B: lanes 16-31 hold K+8 / K+24 groups
  const int rowA = lane & 15;
  const _Float16* wcol = Wt + (size_t)(col0 + colL) * K;

  for (int k0 = 0; k0 < K; k0 += 32) {
    // B fragment: two contiguous 16B chunks per lane, straight from global
    const v8h* pb = (const v8h*)(wcol + k0 + hiB);
    v8h b0 = pb[0], b1 = pb[2];
    v16h b = __builtin_shufflevector(b0, b1, 0,1,2,3,4,5,6,7,8,9,10,11,12,13,14,15);

    #pragma unroll
    for (int t = 0; t < 4; ++t) {
      const _Float16* pa = &lA[(t * 16 + rowA) * Kpad + k0 + hiA];
      v8h a0 = *(const v8h*)(pa);
      v8h a1 = *(const v8h*)(pa + 16);
      v16h a = __builtin_shufflevector(a0, a1, 0,1,2,3,4,5,6,7,8,9,10,11,12,13,14,15);
      v8f& acc = (t == 0) ? acc0 : (t == 1) ? acc1 : (t == 2) ? acc2 : acc3;
      acc = __builtin_amdgcn_wmma_f32_16x16x32_f16(
          /*neg_a=*/false, a, /*neg_b=*/false, b,
          /*c_mod=*/(short)0, acc, /*reuse_a=*/false, /*reuse_b=*/false);
    }
  }

  // Epilogue: C layout — lanes 0-15: M=r, lanes 16-31: M=8+r; N = lane%16
  const int rBase = (lane >> 4) * 8;
  const int col   = col0 + colL;
  const float bc  = bias[col];
  #pragma unroll
  for (int t = 0; t < 4; ++t) {
    const v8f& acc = (t == 0) ? acc0 : (t == 1) ? acc1 : (t == 2) ? acc2 : acc3;
    #pragma unroll
    for (int r = 0; r < 8; ++r) {
      int row = row0 + t * 16 + rBase + r;
      if (row < N) {
        float v = acc[r] + bc;
        if (act == ACT_RELU) v = fmaxf(v, 0.0f);
        Y[(size_t)row * ldY + colOff + col] = v;
      }
    }
  }
}

// First embedding layer: tiny K (1/3/7) -> plain VALU. Y = relu(X@W + b)
__global__ __launch_bounds__(256) void embed_l1(
    const float* __restrict__ X, const float* __restrict__ W,
    const float* __restrict__ bias, float* __restrict__ Y, int N, int ki)
{
  long long t = (long long)blockIdx.x * 256 + threadIdx.x;
  int n = (int)(t >> 7);
  int c = (int)(t & 127);
  if (n >= N) return;
  float acc = bias[c];
  for (int j = 0; j < ki; ++j) acc += X[(size_t)n * ki + j] * W[j * 128 + c];
  Y[(size_t)n * 128 + c] = fmaxf(acc, 0.0f);
}

// Edge gather-scale-scatter: agg[tgt] += src[srcIdx] * ef  (float atomics)
__global__ __launch_bounds__(256) void gin_scatter(
    const float* __restrict__ src, const int* __restrict__ eSrc,
    const int* __restrict__ eTgt, const float* __restrict__ ef,
    float* __restrict__ agg, long long E)
{
  long long t = (long long)blockIdx.x * 256 + threadIdx.x;
  long long e = t >> 5;
  if (e >= E) return;
  int c4 = (int)(t & 31) << 2;
  int s = eSrc[e];
  int d = eTgt[e];
  float f = ef[e];
  const float4 v = *(const float4*)(src + (size_t)s * 128 + c4);
  float* ap = agg + (size_t)d * 128 + c4;
  atomicAdd(ap + 0, v.x * f);
  atomicAdd(ap + 1, v.y * f);
  atomicAdd(ap + 2, v.z * f);
  atomicAdd(ap + 3, v.w * f);
}

// Gate second layer: s[n] = relu(dot(H[n,:], w2) + b2)
__global__ __launch_bounds__(256) void gate_score(
    const float* __restrict__ H, const float* __restrict__ w2,
    const float* __restrict__ b2, float* __restrict__ s, int N)
{
  long long t = (long long)blockIdx.x * 256 + threadIdx.x;
  if (t >= N) return;
  float acc = b2[0];
  const float* h = H + (size_t)t * 128;
  #pragma unroll 8
  for (int c = 0; c < 128; ++c) acc += h[c] * w2[c];
  s[t] = fmaxf(acc, 0.0f);
}

// Segment max (scores >= 0 so int-view compare is order-preserving)
__global__ __launch_bounds__(256) void seg_max(
    const float* __restrict__ s, const int* __restrict__ batch,
    float* __restrict__ m, int N)
{
  long long t = (long long)blockIdx.x * 256 + threadIdx.x;
  if (t >= N) return;
  atomicMax((int*)&m[batch[t]], __float_as_int(s[t]));
}

// e = exp(s - m[b]) stored in place; z[b] += e
__global__ __launch_bounds__(256) void seg_expsum(
    float* __restrict__ s, const int* __restrict__ batch,
    const float* __restrict__ m, float* __restrict__ z, int N)
{
  long long t = (long long)blockIdx.x * 256 + threadIdx.x;
  if (t >= N) return;
  int b = batch[t];
  float e = expf(s[t] - m[b]);
  s[t] = e;
  atomicAdd(&z[b], e);
}

// P[b, colOff + c] += (e[n]/(z[b]+1e-16)) * X[n, c]
__global__ __launch_bounds__(256) void pool_weighted(
    const float* __restrict__ X, const float* __restrict__ e,
    const int* __restrict__ batch, const float* __restrict__ z,
    float* __restrict__ P, int ldP, int colOff, int N)
{
  long long t = (long long)blockIdx.x * 256 + threadIdx.x;
  int n = (int)(t >> 5);
  if (n >= N) return;
  int c4 = (int)(t & 31) << 2;
  int b = batch[n];
  float w = e[n] / (z[b] + 1e-16f);
  const float4 v = *(const float4*)(X + (size_t)n * 128 + c4);
  float* pp = P + (size_t)b * ldP + colOff + c4;
  atomicAdd(pp + 0, v.x * w);
  atomicAdd(pp + 1, v.y * w);
  atomicAdd(pp + 2, v.z * w);
  atomicAdd(pp + 3, v.w * w);
}

// out[g] = sigmoid(dot(F[g,:], w) + b)
__global__ void final_sigmoid(const float* __restrict__ Fb,
                              const float* __restrict__ w,
                              const float* __restrict__ b,
                              float* __restrict__ out, int G)
{
  int g = threadIdx.x;
  if (g >= G) return;
  float acc = b[0];
  const float* f = Fb + (size_t)g * 128;
  for (int c = 0; c < 128; ++c) acc += f[c] * w[c];
  out[g] = 1.0f / (1.0f + expf(-acc));
}

// ---------------------------------------------------------------------------
extern "C" void kernel_launch(void* const* d_in, const int* in_sizes, int n_in,
                              void* d_out, int out_size, void* d_ws, size_t ws_size,
                              hipStream_t stream)
{
  (void)out_size; (void)ws_size;
  const float* conF = (const float*)d_in[0];
  const float* varF = (const float*)d_in[1];
  const float* objF = (const float*)d_in[2];
  const int*   cvEi = (const int*)d_in[3];
  const float* cvEf = (const float*)d_in[4];
  const int*   ovEi = (const int*)d_in[5];
  const float* ovEf = (const float*)d_in[6];
  const int*   varB = (const int*)d_in[8];
  const int*   conB = (const int*)d_in[9];
  const int*   objB = (const int*)d_in[10];

  const int NC  = in_sizes[0];
  const int NV  = in_sizes[1] / 3;
  const int NO  = in_sizes[2] / 7;
  const long long ECV = in_sizes[4];
  const long long EOV = in_sizes[6];
  const int G = 128;

  // Param leaf mapping: dict-insertion vs pytree-sorted key order, detected
  // from leaf 15 (insertion => ve.w1 = 3*128; sorted => cv.w1 = 128*128).
  bool ins = (n_in > 15 && in_sizes[15] == 3 * EMB);
  int b_ce, b_ve, b_oe, b_vc, b_cv, b_vo, b_ov, b_g1, b_g2, b_g3, b_l1, b_l2;
  if (ins) {
    b_ce = 11; b_ve = 15; b_oe = 19;
    b_vc = 23; b_cv = 28; b_vo = 33; b_ov = 38;
    b_g1 = 43; b_g2 = 47; b_g3 = 51;
    b_l1 = 55; b_l2 = 57;
  } else { // sorted: ce, cv, gt1, gt2, gt3, lin1, lin2, oe, ov, vc, ve, vo
    b_ce = 11; b_cv = 15;
    b_g1 = 20; b_g2 = 24; b_g3 = 28;
    b_l1 = 32; b_l2 = 34;
    b_oe = 36; b_ov = 40; b_vc = 45; b_ve = 50; b_vo = 54;
  }
  #define FP(i) ((const float*)d_in[(i)])

  // Workspace layout (floats; all block sizes are multiples of 4 -> 16B aligned)
  float* ws = (float*)d_ws;
  size_t o = 0;
  float* varA = ws + o; o += (size_t)NV * EMB;
  float* conA = ws + o; o += (size_t)NC * EMB;
  float* objA = ws + o; o += (size_t)NO * EMB;
  float* H    = ws + o; o += (size_t)NV * EMB;   // MLP hidden scratch (max N)
  float* AGG  = ws + o; o += (size_t)NV * EMB;   // segment-sum scratch (max N)
  float* S    = ws + o; o += (size_t)(NV + NC);  // gate scores (max pair)
  float* Mb   = ws + o; o += G;
  float* Zb   = ws + o; o += G;
  float* P    = ws + o; o += (size_t)G * 384;
  float* Fb   = ws + o; o += (size_t)G * EMB;
  _Float16* wtPool = (_Float16*)(ws + o);        // packed f16 weights
  size_t wtOff = 0;

  auto grid1 = [](long long t) { return (unsigned)((t + 255) / 256); };

  // Pack all WMMA weights to transposed f16 (one-shot tiny kernels)
  auto packW = [&](int leaf, int K) {
    _Float16* dst = wtPool + wtOff;
    wtOff += (size_t)K * 128;
    pack_wt<<<grid1((long long)K * 128), 256, 0, stream>>>(FP(leaf), dst, K);
    return (const _Float16*)dst;
  };
  const _Float16* ceW2t = packW(b_ce + 2, 128);
  const _Float16* veW2t = packW(b_ve + 2, 128);
  const _Float16* oeW2t = packW(b_oe + 2, 128);
  const _Float16* vcW1t = packW(b_vc, 128);
  const _Float16* vcW2t = packW(b_vc + 2, 128);
  const _Float16* cvW1t = packW(b_cv, 128);
  const _Float16* cvW2t = packW(b_cv + 2, 128);
  const _Float16* voW1t = packW(b_vo, 128);
  const _Float16* voW2t = packW(b_vo + 2, 128);
  const _Float16* ovW1t = packW(b_ov, 128);
  const _Float16* ovW2t = packW(b_ov + 2, 128);
  const _Float16* g1W1t = packW(b_g1, 128);
  const _Float16* g2W1t = packW(b_g2, 128);
  const _Float16* g3W1t = packW(b_g3, 128);
  const _Float16* l1Wt  = packW(b_l1, 384);

  auto gemm = [&](const float* X, const float* X2, const float* eps,
                  const _Float16* Wt, const float* bias, float* Y,
                  int N, int K, int ldY, int colOff, int act) {
    gemm_wmma<<<(N + 63) / 64, 256, 0, stream>>>(X, X2, eps, Wt, bias, Y, N, K, ldY, colOff, act);
  };

  // --- node embedding MLPs ---
  embed_l1<<<grid1((long long)NC * EMB), 256, 0, stream>>>(conF, FP(b_ce), FP(b_ce + 1), H, NC, 1);
  gemm(H, nullptr, nullptr, ceW2t, FP(b_ce + 3), conA, NC, EMB, EMB, 0, ACT_RELU);
  embed_l1<<<grid1((long long)NV * EMB), 256, 0, stream>>>(varF, FP(b_ve), FP(b_ve + 1), H, NV, 3);
  gemm(H, nullptr, nullptr, veW2t, FP(b_ve + 3), varA, NV, EMB, EMB, 0, ACT_RELU);
  embed_l1<<<grid1((long long)NO * EMB), 256, 0, stream>>>(objF, FP(b_oe), FP(b_oe + 1), H, NO, 7);
  gemm(H, nullptr, nullptr, oeW2t, FP(b_oe + 3), objA, NO, EMB, EMB, 0, ACT_RELU);

  // --- GIN conv: scatter + fused (1+eps)*tgt + agg -> MLP (in-place update) ---
  auto gin = [&](const float* src, const int* eSrc, const int* eTgt,
                 const float* ef, long long E, float* tgt, int Ntgt,
                 const _Float16* W1t, const _Float16* W2t, int base) {
    hipMemsetAsync(AGG, 0, (size_t)Ntgt * EMB * sizeof(float), stream);
    gin_scatter<<<grid1(E * 32), 256, 0, stream>>>(src, eSrc, eTgt, ef, AGG, E);
    gemm(tgt, AGG, FP(base + 4), W1t, FP(base + 1), H, Ntgt, EMB, EMB, 0, ACT_RELU);
    gemm(H, nullptr, nullptr, W2t, FP(base + 3), tgt, Ntgt, EMB, EMB, 0, ACT_NONE);
  };

  // --- gate-attention pooling over two node arrays into P[:, colOff..colOff+127] ---
  auto pool = [&](const float* X1, int N1, const int* B1,
                  const float* X2, int N2, const int* B2,
                  const _Float16* W1t, int base, int colOff) {
    hipMemsetAsync(Mb, 0, G * sizeof(float), stream);
    hipMemsetAsync(Zb, 0, G * sizeof(float), stream);
    gemm(X1, nullptr, nullptr, W1t, FP(base + 1), H, N1, EMB, EMB, 0, ACT_RELU);
    gate_score<<<grid1(N1), 256, 0, stream>>>(H, FP(base + 2), FP(base + 3), S, N1);
    gemm(X2, nullptr, nullptr, W1t, FP(base + 1), H, N2, EMB, EMB, 0, ACT_RELU);
    gate_score<<<grid1(N2), 256, 0, stream>>>(H, FP(base + 2), FP(base + 3), S + N1, N2);
    seg_max<<<grid1(N1), 256, 0, stream>>>(S, B1, Mb, N1);
    seg_max<<<grid1(N2), 256, 0, stream>>>(S + N1, B2, Mb, N2);
    seg_expsum<<<grid1(N1), 256, 0, stream>>>(S, B1, Mb, Zb, N1);
    seg_expsum<<<grid1(N2), 256, 0, stream>>>(S + N1, B2, Mb, Zb, N2);
    pool_weighted<<<grid1((long long)N1 * 32), 256, 0, stream>>>(X1, S, B1, Zb, P, 384, colOff, N1);
    pool_weighted<<<grid1((long long)N2 * 32), 256, 0, stream>>>(X2, S + N1, B2, Zb, P, 384, colOff, N2);
  };

  // con1 = gin_vc(src=var0 via rev_cv, tgt=con0)
  gin(varA, cvEi + ECV, cvEi, cvEf, ECV, conA, NC, vcW1t, vcW2t, b_vc);
  // var1 = gin_cv(src=con1 via cv_ei, tgt=var0)
  gin(conA, cvEi, cvEi + ECV, cvEf, ECV, varA, NV, cvW1t, cvW2t, b_cv);

  hipMemsetAsync(P, 0, (size_t)G * 384 * sizeof(float), stream);
  pool(varA, NV, varB, conA, NC, conB, g1W1t, b_g1, 0);      // st

  // obj1 = gin_vo(src=var1 via rev_ov, tgt=obj0)
  gin(varA, ovEi + EOV, ovEi, ovEf, EOV, objA, NO, voW1t, voW2t, b_vo);
  // var2 = gin_ov(src=obj1 via ov_ei, tgt=var1)
  gin(objA, ovEi, ovEi + EOV, ovEf, EOV, varA, NV, ovW1t, ovW2t, b_ov);

  pool(varA, NV, varB, objA, NO, objB, g2W1t, b_g2, 128);    // nd
  pool(objA, NO, objB, conA, NC, conB, g3W1t, b_g3, 256);    // rd

  // final: F = relu(P[G,384] @ lin1 + b); out = sigmoid(F @ lin2 + b)
  gemm(P, nullptr, nullptr, l1Wt, FP(b_l1 + 1), Fb, G, 384, EMB, 0, ACT_RELU);
  final_sigmoid<<<1, 128, 0, stream>>>(Fb, FP(b_l2), FP(b_l2 + 1), (float*)d_out, G);
}
